// MaskedHeteroGAT_29523605193036
// MI455X (gfx1250) — matrix operand order repaired and checked by reading, not compiled, not run on hardware
//
#include <hip/hip_runtime.h>

typedef __bf16 bf16;
typedef __attribute__((ext_vector_type(16))) __bf16 v16bf;
typedef __attribute__((ext_vector_type(8)))  float  v8f;

#define NPKG   10000
#define NTGT   20000
#define NEDGE  100000
#define FIN    400
#define DM     256
#define NHEAD  4
#define CHEAD  64
#define NREL   5
#define NCLU   64
#define NEGS   0.2f

// ---------- helpers ----------
__device__ __forceinline__ bf16 f2bf(float f) {
    unsigned u = __float_as_uint(f);
    unsigned r = (u + 0x7FFFu + ((u >> 16) & 1u)) >> 16;   // round-to-nearest-even
    unsigned short h = (unsigned short)r;
    return __builtin_bit_cast(bf16, h);
}
__device__ __forceinline__ unsigned short f2bfraw(float f) {
    unsigned u = __float_as_uint(f);
    return (unsigned short)((u + 0x7FFFu + ((u >> 16) & 1u)) >> 16);
}
__device__ __forceinline__ unsigned encf(float f) {        // order-preserving f32->u32
    unsigned u = __float_as_uint(f);
    return (u & 0x80000000u) ? ~u : (u | 0x80000000u);
}
__device__ __forceinline__ float decf(unsigned u) {
    return (u & 0x80000000u) ? __uint_as_float(u ^ 0x80000000u) : __uint_as_float(~u);
}

// ---------- elementwise utility kernels ----------
__global__ void k_cvt_bf16(const float* __restrict__ src, bf16* __restrict__ dst, long n) {
    long i = (long)blockIdx.x * blockDim.x + threadIdx.x;
    if (i < n) dst[i] = f2bf(src[i]);
}
__global__ void k_fill_f32(float* p, float v, long n) {
    long i = (long)blockIdx.x * blockDim.x + threadIdx.x;
    if (i < n) p[i] = v;
}
__global__ void k_fill_u32(unsigned* p, unsigned v, long n) {
    long i = (long)blockIdx.x * blockDim.x + threadIdx.x;
    if (i < n) p[i] = v;
}
__global__ void k_relu_ip(float* p, long n) {
    long i = (long)blockIdx.x * blockDim.x + threadIdx.x;
    if (i < n) p[i] = fmaxf(p[i], 0.0f);
}
// sT[c][n] = (bf16) s[n][c]
__global__ void k_transpose_bf16(const float* __restrict__ s, bf16* __restrict__ sT, int nrows, int ncols) {
    long i = (long)blockIdx.x * blockDim.x + threadIdx.x;
    if (i >= (long)nrows * ncols) return;
    int r = (int)(i / ncols), c = (int)(i % ncols);
    sT[(size_t)c * nrows + r] = f2bf(s[i]);
}

// ---------- pack f32 B[K,N] into WMMA-fragment-ready bf16 dwords ----------
// dword o = ((nt*KT + kt)*32 + lane)*8 + j  holds pair (B[k][n], B[k+1][n]),
// k = kt*32 + (lane>>4)*8 + (j&3)*2 + (j>>2)*16, n = nt*16 + (lane&15).  OOB k -> 0.
__global__ void k_pack_b(const float* __restrict__ S, unsigned* __restrict__ P, int K, int N) {
    int KT = (K + 31) >> 5, NT = N >> 4;
    long total = (long)KT * NT * 256;
    long o = (long)blockIdx.x * blockDim.x + threadIdx.x;
    if (o >= total) return;
    int j = (int)(o & 7);
    int lane = (int)((o >> 3) & 31);
    long blk = o >> 8;
    int kt = (int)(blk % KT), nt = (int)(blk / KT);
    int k = kt * 32 + (lane >> 4) * 8 + (j & 3) * 2 + (j >> 2) * 16;
    int n = nt * 16 + (lane & 15);
    unsigned lo = 0, hi = 0;
    if (k < K)     lo = (unsigned)f2bfraw(S[(size_t)k * N + n]);
    if (k + 1 < K) hi = (unsigned)f2bfraw(S[(size_t)(k + 1) * N + n]);
    P[o] = lo | (hi << 16);
}

// ---------- WMMA bf16 GEMM: C[M,N] = act(A[M,K] @ B[K,N]) ----------
// A row-major bf16, Bp packed by k_pack_b, C f32. One wave per 16x64 C tile.
// Requires M%16==0, N%64==0, K%16==0 (true for all call sites).
__global__ __launch_bounds__(32)
void k_gemm_bf16(const bf16* __restrict__ A, const unsigned* __restrict__ Bp,
                 float* __restrict__ C, int M, int N, int K, int act) {
    const int lane = threadIdx.x;                 // wave32
    const int row0 = blockIdx.x * 16;
    const int nt0  = blockIdx.y * 4;              // 4 N-subtiles of 16
    const int rr = lane & 15, gg = lane >> 4;
    const int KT = (K + 31) >> 5;
    const int KFULL = K >> 5;                     // number of complete 32-wide K tiles

    v8f acc[4] = {};
    const bf16* Arow = A + (size_t)(row0 + rr) * K + gg * 8;
    const unsigned* Bq[4];
    #pragma unroll
    for (int t = 0; t < 4; ++t)
        Bq[t] = Bp + ((size_t)(nt0 + t) * KT) * 256 + lane * 8;

    // steady state: no bounds checks, pure load+WMMA
    for (int kt = 0; kt < KFULL; ++kt) {
        v16bf af;
        uint4* afv = reinterpret_cast<uint4*>(&af);
        afv[0] = *reinterpret_cast<const uint4*>(Arow + kt * 32);
        afv[1] = *reinterpret_cast<const uint4*>(Arow + kt * 32 + 16);
        #pragma unroll
        for (int t = 0; t < 4; ++t) {
            const uint4* bp = reinterpret_cast<const uint4*>(Bq[t] + (size_t)kt * 256);
            v16bf bfv;
            uint4* bv = reinterpret_cast<uint4*>(&bfv);
            bv[0] = bp[0]; bv[1] = bp[1];
            acc[t] = __builtin_amdgcn_wmma_f32_16x16x32_bf16(false, af, false, bfv,
                                                             (short)0, acc[t], false, false);
        }
    }
    // tail (K % 32 == 16): upper half of A fragment is zero; packed B already zero-padded
    if (KFULL != KT) {
        v16bf af;
        uint4* afv = reinterpret_cast<uint4*>(&af);
        afv[0] = *reinterpret_cast<const uint4*>(Arow + KFULL * 32);
        uint4 z; z.x = z.y = z.z = z.w = 0u;
        afv[1] = z;
        #pragma unroll
        for (int t = 0; t < 4; ++t) {
            const uint4* bp = reinterpret_cast<const uint4*>(Bq[t] + (size_t)KFULL * 256);
            v16bf bfv;
            uint4* bv = reinterpret_cast<uint4*>(&bfv);
            bv[0] = bp[0]; bv[1] = bp[1];
            acc[t] = __builtin_amdgcn_wmma_f32_16x16x32_bf16(false, af, false, bfv,
                                                             (short)0, acc[t], false, false);
        }
    }
    // C/D layout: VGPR v holds M = v + gg*8, N = rr (per 16-col subtile)
    #pragma unroll
    for (int t = 0; t < 4; ++t) {
        #pragma unroll
        for (int v = 0; v < 8; ++v) {
            float val = acc[t][v];
            if (act) val = fmaxf(val, 0.0f);
            C[(size_t)(row0 + v + gg * 8) * N + (nt0 + t) * 16 + rr] = val;
        }
    }
}

// ---------- row softmax over 64 columns (one wave per row) ----------
__global__ __launch_bounds__(32)
void k_softmax64(float* s) {
    float* p = s + (size_t)blockIdx.x * 64;
    int lane = threadIdx.x;
    float a = p[lane], b = p[lane + 32];
    float m = fmaxf(a, b);
    for (int o = 16; o; o >>= 1) m = fmaxf(m, __shfl_xor(m, o, 32));
    a = __expf(a - m); b = __expf(b - m);
    float t = a + b;
    for (int o = 16; o; o >>= 1) t += __shfl_xor(t, o, 32);
    p[lane] = a / t; p[lane + 32] = b / t;
}

// ---------- conv1 edge phase ----------
__global__ void k_edge_logit(const float* __restrict__ hs, const float* __restrict__ ht,
                             const float* __restrict__ a1, const int* __restrict__ si,
                             const int* __restrict__ di, float* __restrict__ logit,
                             unsigned* __restrict__ segmax) {
    int idx = blockIdx.x * blockDim.x + threadIdx.x;
    if (idx >= NEDGE * NHEAD) return;
    int e = idx >> 2, h = idx & 3;
    int s = si[e], d = di[e];
    const float* ps = hs + (size_t)s * DM + h * CHEAD;
    const float* pt = ht + (size_t)d * DM + h * CHEAD;
    const float* pa = a1 + h * CHEAD;
    float acc = 0.0f;
    #pragma unroll 4
    for (int c = 0; c < CHEAD; ++c) {
        float z = ps[c] + pt[c];
        z = z > 0.0f ? z : NEGS * z;
        acc += z * pa[c];
    }
    logit[idx] = acc;
    atomicMax(&segmax[d * NHEAD + h], encf(acc));
}
__global__ void k_edge_expsum(float* __restrict__ logit, const unsigned* __restrict__ segmax,
                              float* __restrict__ segden, const int* __restrict__ di) {
    int idx = blockIdx.x * blockDim.x + threadIdx.x;
    if (idx >= NEDGE * NHEAD) return;
    int e = idx >> 2, h = idx & 3;
    int d = di[e];
    float ex = __expf(logit[idx] - decf(segmax[d * NHEAD + h]));
    logit[idx] = ex;
    atomicAdd(&segden[d * NHEAD + h], ex);
}
__global__ __launch_bounds__(256)
void k_edge_scatter(const float* __restrict__ ex, const float* __restrict__ den,
                    const float* __restrict__ hs, const int* __restrict__ si,
                    const int* __restrict__ di, float* __restrict__ outp) {
    int e = blockIdx.x, f = threadIdx.x;     // 256 channels
    int h = f >> 6;
    int s = si[e], d = di[e];
    float alpha = ex[e * NHEAD + h] / den[d * NHEAD + h];
    atomicAdd(&outp[(size_t)d * DM + f], alpha * hs[(size_t)s * DM + f]);
}

// ---------- loss accumulators ----------
__global__ void k_ent_accum(const float* __restrict__ s, long n, float* acc) {
    __shared__ float red[256];
    long i = (long)blockIdx.x * blockDim.x + threadIdx.x;
    long stride = (long)gridDim.x * blockDim.x;
    float v = 0.0f;
    for (; i < n; i += stride) { float x = s[i]; v += x * logf(x + 1e-15f); }
    red[threadIdx.x] = v; __syncthreads();
    for (int o = 128; o; o >>= 1) { if ((int)threadIdx.x < o) red[threadIdx.x] += red[threadIdx.x + o]; __syncthreads(); }
    if (threadIdx.x == 0) atomicAdd(acc, red[0]);
}
__global__ void k_dot_accum(const float* __restrict__ a, const float* __restrict__ b, long n, float* acc) {
    __shared__ float red[256];
    long i = (long)blockIdx.x * blockDim.x + threadIdx.x;
    long stride = (long)gridDim.x * blockDim.x;
    float v = 0.0f;
    for (; i < n; i += stride) v += a[i] * b[i];
    red[threadIdx.x] = v; __syncthreads();
    for (int o = 128; o; o >>= 1) { if ((int)threadIdx.x < o) red[threadIdx.x] += red[threadIdx.x + o]; __syncthreads(); }
    if (threadIdx.x == 0) atomicAdd(acc, red[0]);
}
__global__ void k_cross_accum(const float* __restrict__ sp, const float* __restrict__ st,
                              const int* __restrict__ si, const int* __restrict__ di, float* acc) {
    __shared__ float red[256];
    int e = blockIdx.x * blockDim.x + threadIdx.x;
    float v = 0.0f;
    if (e < NEDGE) {
        const float* a = sp + (size_t)si[e] * NCLU;
        const float* b = st + (size_t)di[e] * NCLU;
        #pragma unroll 4
        for (int c = 0; c < NCLU; ++c) v += a[c] * b[c];
    }
    red[threadIdx.x] = v; __syncthreads();
    for (int o = 128; o; o >>= 1) { if ((int)threadIdx.x < o) red[threadIdx.x] += red[threadIdx.x + o]; __syncthreads(); }
    if (threadIdx.x == 0) atomicAdd(acc, red[0]);
}

// ---------- conv2 dense attention: alpha over 64 sources per (j,h) ----------
__global__ __launch_bounds__(32)
void k_dense_att(const float* __restrict__ hs2, const float* __restrict__ ht2,
                 const float* __restrict__ a2, float* __restrict__ alpha_out) {
    int j = blockIdx.x, h = blockIdx.y, lane = threadIdx.x;
    const float* pt = ht2 + (size_t)j * DM + h * CHEAD;
    const float* pa = a2 + h * CHEAD;
    const float* p0 = hs2 + (size_t)lane * DM + h * CHEAD;
    const float* p1 = hs2 + (size_t)(lane + 32) * DM + h * CHEAD;
    float l0 = 0.0f, l1 = 0.0f;
    #pragma unroll 4
    for (int c = 0; c < CHEAD; ++c) {
        float t = pt[c], ac = pa[c];
        float z0 = p0[c] + t; z0 = z0 > 0.0f ? z0 : NEGS * z0;
        float z1 = p1[c] + t; z1 = z1 > 0.0f ? z1 : NEGS * z1;
        l0 += z0 * ac; l1 += z1 * ac;
    }
    float m = fmaxf(l0, l1);
    for (int o = 16; o; o >>= 1) m = fmaxf(m, __shfl_xor(m, o, 32));
    float e0 = __expf(l0 - m), e1 = __expf(l1 - m);
    float t = e0 + e1;
    for (int o = 16; o; o >>= 1) t += __shfl_xor(t, o, 32);
    alpha_out[(size_t)lane * (NCLU * NHEAD) + j * NHEAD + h]        = e0 / t;  // [i][j][h]
    alpha_out[(size_t)(lane + 32) * (NCLU * NHEAD) + j * NHEAD + h] = e1 / t;
}

// ---------- classifier + loss finalize ----------
__global__ void k_classifier(const float* __restrict__ pooled, const float* __restrict__ Wc,
                             const float* __restrict__ bc, float* __restrict__ out) {
    int row = blockIdx.x * blockDim.x + threadIdx.x;
    if (row >= 6 * NCLU) return;
    const float* pr = pooled + (size_t)row * DM;
    float acc = bc[0];
    #pragma unroll 4
    for (int i = 0; i < DM; ++i) acc += pr[i] * Wc[i];
    out[row] = 1.0f / (1.0f + __expf(-acc));
}
__global__ void k_finalize(const float* __restrict__ sc, float* __restrict__ out) {
    float loss = 0.0f;
    for (int r = 0; r < NREL; ++r) {
        float sq = (float)NEDGE - 2.0f * sc[6 + r] + sc[11 + r];
        sq = fmaxf(sq, 0.0f);
        loss += sqrtf(sq) / ((float)NPKG * (float)NTGT);
    }
    loss += -sc[0] / (float)NPKG;
    for (int r = 0; r < NREL; ++r) loss += -sc[1 + r] / (float)NTGT;
    out[0] = loss;
}

// ---------- host ----------
static inline char* ws_take(char*& p, size_t bytes) {
    char* r = p;
    p += (bytes + 255) & ~(size_t)255;
    return r;
}
static inline unsigned blocks(long n, int bs) { return (unsigned)((n + bs - 1) / bs); }
static inline long packed_dw(int K, int N) { return (long)((K + 31) / 32) * (N / 16) * 256; }

extern "C" void kernel_launch(void* const* d_in, const int* in_sizes, int n_in,
                              void* d_out, int out_size, void* d_ws, size_t ws_size,
                              hipStream_t stream) {
    (void)in_sizes; (void)n_in; (void)out_size; (void)ws_size;
    const float* x_pkg  = (const float*)d_in[0];
    const float* x_tgt  = (const float*)d_in[1];
    const float* W1_src = (const float*)d_in[2];
    const float* W1_dst = (const float*)d_in[3];
    const float* a1     = (const float*)d_in[4];
    const float* W_pkg  = (const float*)d_in[5];
    const float* Ws1    = (const float*)d_in[6];
    const float* Ws2    = (const float*)d_in[7];
    const float* W2_src = (const float*)d_in[8];
    const float* W2_dst = (const float*)d_in[9];
    const float* a2     = (const float*)d_in[10];
    const float* Wc     = (const float*)d_in[11];
    const float* bc     = (const float*)d_in[12];
    const int*   src_i  = (const int*)d_in[13];
    const int*   dst_i  = (const int*)d_in[14];
    float* out = (float*)d_out;

    const long DW_W1  = packed_dw(FIN, DM);    // per-relation W1 packed dwords
    const long DW_W2  = packed_dw(DM, DM);     // per-relation W2 packed dwords

    // workspace layout (per-relation buffers reused)
    char* p = (char*)d_ws;
    bf16* xpkg_b  = (bf16*)ws_take(p, (size_t)NPKG * FIN * 2);         // A roles
    bf16* xtgt_b  = (bf16*)ws_take(p, (size_t)NTGT * FIN * 2);
    bf16* hpkg_b  = (bf16*)ws_take(p, (size_t)NPKG * DM * 2);
    bf16* htgt_b  = (bf16*)ws_take(p, (size_t)NTGT * DM * 2);
    bf16* asg1_b  = (bf16*)ws_take(p, (size_t)NTGT * 128 * 2);
    bf16* spkgT_b = (bf16*)ws_take(p, (size_t)NPKG * NCLU * 2);
    bf16* stgtT_b = (bf16*)ws_take(p, (size_t)NTGT * NCLU * 2);
    bf16* pooled_b= (bf16*)ws_take(p, (size_t)6 * NCLU * DM * 2);
    unsigned* wpkgP = (unsigned*)ws_take(p, (size_t)packed_dw(FIN, DM) * 4);   // B roles (packed)
    unsigned* ws1P  = (unsigned*)ws_take(p, (size_t)packed_dw(DM, 128) * 4);
    unsigned* ws2P  = (unsigned*)ws_take(p, (size_t)packed_dw(128, 64) * 4);
    unsigned* w1sP  = (unsigned*)ws_take(p, (size_t)NREL * DW_W1 * 4);
    unsigned* w1dP  = (unsigned*)ws_take(p, (size_t)NREL * DW_W1 * 4);
    unsigned* w2sP  = (unsigned*)ws_take(p, (size_t)NREL * DW_W2 * 4);
    unsigned* w2dP  = (unsigned*)ws_take(p, (size_t)NREL * DW_W2 * 4);
    unsigned* hpkgP = (unsigned*)ws_take(p, (size_t)packed_dw(NPKG, DM) * 4);
    unsigned* htgtP = (unsigned*)ws_take(p, (size_t)packed_dw(NTGT, DM) * 4);
    unsigned* spkgP = (unsigned*)ws_take(p, (size_t)packed_dw(NPKG, NCLU) * 4);
    unsigned* stgtP = (unsigned*)ws_take(p, (size_t)packed_dw(NTGT, NCLU) * 4);
    float* hs1    = (float*)ws_take(p, (size_t)NPKG * DM * 4);
    float* ht1    = (float*)ws_take(p, (size_t)NTGT * DM * 4);
    float* htgt   = (float*)ws_take(p, (size_t)NTGT * DM * 4);
    float* hpkg   = (float*)ws_take(p, (size_t)NPKG * DM * 4);
    float* logitb = (float*)ws_take(p, (size_t)NEDGE * NHEAD * 4);
    unsigned* smax= (unsigned*)ws_take(p, (size_t)NTGT * NHEAD * 4);
    float* sden   = (float*)ws_take(p, (size_t)NTGT * NHEAD * 4);
    float* asg1   = (float*)ws_take(p, (size_t)NTGT * 128 * 4);
    float* spkg   = (float*)ws_take(p, (size_t)NPKG * NCLU * 4);
    float* stgt   = (float*)ws_take(p, (size_t)NTGT * NCLU * 4);
    float* pooled = (float*)ws_take(p, (size_t)6 * NCLU * DM * 4);
    float* Gp     = (float*)ws_take(p, (size_t)NCLU * NCLU * 4);
    float* Tt     = (float*)ws_take(p, (size_t)NCLU * NCLU * 4);
    float* hs2    = (float*)ws_take(p, (size_t)NCLU * DM * 4);
    float* ht2    = (float*)ws_take(p, (size_t)NCLU * DM * 4);
    float* scal   = (float*)ws_take(p, 64 * 4);  // [0]=entPkg [1..5]=entTgt [6..10]=cross [11..15]=GpTt

    auto cvt = [&](const float* s, bf16* d, long n) {
        k_cvt_bf16<<<blocks(n, 256), 256, 0, stream>>>(s, d, n);
    };
    auto pack = [&](const float* s, unsigned* d, int K, int N) {
        k_pack_b<<<blocks(packed_dw(K, N), 256), 256, 0, stream>>>(s, d, K, N);
    };
    auto gemm = [&](const bf16* A, const unsigned* Bp, float* C, int M, int N, int K, int act) {
        k_gemm_bf16<<<dim3((unsigned)(M / 16), (unsigned)(N / 64)), 32, 0, stream>>>(A, Bp, C, M, N, K, act);
    };

    // zero accumulators
    k_fill_f32<<<1, 64, 0, stream>>>(scal, 0.0f, 16);

    // one-time conversions: activations (A role) + packed weights (B role)
    cvt(x_pkg, xpkg_b, (long)NPKG * FIN);
    pack(W_pkg, wpkgP, FIN, DM);
    pack(Ws1,   ws1P,  DM, 128);
    pack(Ws2,   ws2P,  128, 64);
    for (int r = 0; r < NREL; ++r) {
        pack(W1_src + (size_t)r * FIN * DM, w1sP + (size_t)r * DW_W1, FIN, DM);
        pack(W1_dst + (size_t)r * FIN * DM, w1dP + (size_t)r * DW_W1, FIN, DM);
        pack(W2_src + (size_t)r * DM * DM,  w2sP + (size_t)r * DW_W2, DM, DM);
        pack(W2_dst + (size_t)r * DM * DM,  w2dP + (size_t)r * DW_W2, DM, DM);
    }

    // ---- pkg side: h_pkg, assignment, pooling, Gp, entropy ----
    gemm(xpkg_b, wpkgP, hpkg, NPKG, DM, FIN, 1);
    cvt(hpkg, hpkg_b, (long)NPKG * DM);
    pack(hpkg, hpkgP, NPKG, DM);
    gemm(hpkg_b, ws1P, asg1, NPKG, 128, DM, 1);
    cvt(asg1, asg1_b, (long)NPKG * 128);
    gemm(asg1_b, ws2P, spkg, NPKG, NCLU, 128, 0);
    k_softmax64<<<NPKG, 32, 0, stream>>>(spkg);
    pack(spkg, spkgP, NPKG, NCLU);
    k_transpose_bf16<<<blocks((long)NPKG * NCLU, 256), 256, 0, stream>>>(spkg, spkgT_b, NPKG, NCLU);
    gemm(spkgT_b, hpkgP, pooled, NCLU, DM, NPKG, 0);           // pooled_pkg
    gemm(spkgT_b, spkgP, Gp, NCLU, NCLU, NPKG, 0);             // Gp = s_pkg^T s_pkg
    k_ent_accum<<<512, 256, 0, stream>>>(spkg, (long)NPKG * NCLU, &scal[0]);

    // ---- per-relation: conv1 GAT + assignment + pooling + loss terms ----
    for (int r = 0; r < NREL; ++r) {
        const int* si = src_i + (size_t)r * NEDGE;
        const int* di = dst_i + (size_t)r * NEDGE;
        cvt(x_tgt + (size_t)r * NTGT * FIN, xtgt_b, (long)NTGT * FIN);
        gemm(xpkg_b, w1sP + (size_t)r * DW_W1, hs1, NPKG, DM, FIN, 0);
        gemm(xtgt_b, w1dP + (size_t)r * DW_W1, ht1, NTGT, DM, FIN, 0);

        k_fill_u32<<<blocks((long)NTGT * NHEAD, 256), 256, 0, stream>>>(smax, 0u, (long)NTGT * NHEAD);
        k_fill_f32<<<blocks((long)NTGT * NHEAD, 256), 256, 0, stream>>>(sden, 0.0f, (long)NTGT * NHEAD);
        k_fill_f32<<<blocks((long)NTGT * DM, 256), 256, 0, stream>>>(htgt, 0.0f, (long)NTGT * DM);

        k_edge_logit<<<blocks((long)NEDGE * NHEAD, 256), 256, 0, stream>>>(
            hs1, ht1, a1 + (size_t)r * NHEAD * CHEAD, si, di, logitb, smax);
        k_edge_expsum<<<blocks((long)NEDGE * NHEAD, 256), 256, 0, stream>>>(logitb, smax, sden, di);
        k_edge_scatter<<<NEDGE, 256, 0, stream>>>(logitb, sden, hs1, si, di, htgt);
        k_relu_ip<<<blocks((long)NTGT * DM, 256), 256, 0, stream>>>(htgt, (long)NTGT * DM);

        cvt(htgt, htgt_b, (long)NTGT * DM);
        pack(htgt, htgtP, NTGT, DM);
        gemm(htgt_b, ws1P, asg1, NTGT, 128, DM, 1);
        cvt(asg1, asg1_b, (long)NTGT * 128);
        gemm(asg1_b, ws2P, stgt, NTGT, NCLU, 128, 0);
        k_softmax64<<<NTGT, 32, 0, stream>>>(stgt);
        pack(stgt, stgtP, NTGT, NCLU);
        k_transpose_bf16<<<blocks((long)NTGT * NCLU, 256), 256, 0, stream>>>(stgt, stgtT_b, NTGT, NCLU);

        gemm(stgtT_b, htgtP, pooled + (size_t)(1 + r) * NCLU * DM, NCLU, DM, NTGT, 0);  // pooled_tgt[r]
        gemm(stgtT_b, stgtP, Tt, NCLU, NCLU, NTGT, 0);                                   // st^T st
        k_dot_accum<<<16, 256, 0, stream>>>(Gp, Tt, (long)NCLU * NCLU, &scal[11 + r]);
        k_cross_accum<<<blocks((long)NEDGE, 256), 256, 0, stream>>>(spkg, stgt, si, di, &scal[6 + r]);
        k_ent_accum<<<512, 256, 0, stream>>>(stgt, (long)NTGT * NCLU, &scal[1 + r]);
    }

    // ---- conv2 dense GAT -> alpha2, classifier, loss ----
    cvt(pooled, pooled_b, (long)6 * NCLU * DM);
    for (int r = 0; r < NREL; ++r) {
        gemm(pooled_b, w2sP + (size_t)r * DW_W2, hs2, NCLU, DM, DM, 0);
        gemm(pooled_b + (size_t)(1 + r) * NCLU * DM, w2dP + (size_t)r * DW_W2, ht2, NCLU, DM, DM, 0);
        k_dense_att<<<dim3(NCLU, NHEAD), 32, 0, stream>>>(
            hs2, ht2, a2 + (size_t)r * NHEAD * CHEAD,
            out + 385 + (size_t)r * NCLU * NCLU * NHEAD);
    }
    k_classifier<<<3, 128, 0, stream>>>(pooled, Wc, bc, out);   // out[0..383]
    k_finalize<<<1, 1, 0, stream>>>(scal, out + 384);           // out[384]
}
